// SE3MessagePassing_36163624632807
// MI455X (gfx1250) — matrix operand order repaired and checked by reading, not compiled, not run on hardware
//
#include <hip/hip_runtime.h>
#include <hip/hip_bf16.h>
#include <math.h>

typedef __attribute__((ext_vector_type(16))) __bf16 v16bf;
typedef __attribute__((ext_vector_type(8)))  float  v8f;

#define HID    128
#define NRBF   16
#define CUTF   5.0f
#define WAVES  4

// ---- shared memory layout (bytes) ----
// W1 (288x128 bf16, fragment-swizzled): 73728
// W2 (128x128): 32768 ; Wc1: 32768
// vec area (b1,b2,bc1,Wa,Wc2 f32 + ba): 2624
// per-wave: bufA 16x288 bf16 (9216) + mbuf 16x128 bf16 (4096) + meta (640)
#define OFF_W1    0
#define OFF_W2    73728
#define OFF_WC1   106496
#define OFF_VEC   139264
#define OFF_WAVE  141888
#define PER_WAVE  13952
#define OFF_BUFA  0
#define OFF_MBUF  9216
#define OFF_META  13312
#define SMEM_TOTAL (OFF_WAVE + WAVES*PER_WAVE)   // 197696 B < 320KB/WGP

// A-matrix 16x32 bf16 fragment addressing (contiguous k order, same as b_idx):
// lane L: row M=L&15, half hi=L>>4; elem j: k = k0 + hi*16 + j
__device__ __forceinline__ int a_idx(int m, int k) {
  int s = k >> 5, kk = k & 31;
  int hi = kk >> 4, w = kk & 15;
  return s*512 + (m + hi*16)*16 + w;
}
// B-matrix 32x16 bf16 fragment addressing:
// lane L: col N=n0+(L&15); elem j: k = k0 + (L>>4)*16 + j
__device__ __forceinline__ int b_idx(int k, int n) {
  int s = k >> 5, kk = k & 31;
  int hi = kk >> 4, j = kk & 15;
  return (s*8 + (n >> 4))*512 + ((n & 15) + hi*16)*16 + j;
}
__device__ __forceinline__ float fsilu(float x){ return x / (1.f + __expf(-x)); }
__device__ __forceinline__ float fsig (float x){ return 1.f / (1.f + __expf(-x)); }

__global__ void egnn_zero_kernel(float* out, long n, float* deg, int n_nodes){
  long i = (long)blockIdx.x*blockDim.x + threadIdx.x;
  long stride = (long)gridDim.x*blockDim.x;
  for (long j=i; j<n; j+=stride) out[j] = 0.f;
  for (long j=i; j<n_nodes; j+=stride) deg[j] = 0.f;
}

__global__ void egnn_degree_kernel(const int* __restrict__ ei, float* deg, int n_edges){
  int i = blockIdx.x*blockDim.x + threadIdx.x;
  if (i < n_edges) atomicAdd(&deg[ei[n_edges + i]], 1.f);
}

__global__ void __launch_bounds__(128)
egnn_edge_kernel(const float* __restrict__ h, const float* __restrict__ pos,
                 const int* __restrict__ ei,
                 const float* __restrict__ W1g, const float* __restrict__ b1g,
                 const float* __restrict__ W2g, const float* __restrict__ b2g,
                 const float* __restrict__ Wag, const float* __restrict__ bag,
                 const float* __restrict__ Wc1g, const float* __restrict__ bc1g,
                 const float* __restrict__ Wc2g,
                 float* __restrict__ hsum, float* __restrict__ psum,
                 int n_nodes, int n_edges)
{
  extern __shared__ char smem[];
  __bf16* W1L  = (__bf16*)(smem + OFF_W1);
  __bf16* W2L  = (__bf16*)(smem + OFF_W2);
  __bf16* Wc1L = (__bf16*)(smem + OFF_WC1);
  float*  vecL = (float*)(smem + OFF_VEC);
  float *b1L=vecL, *b2L=vecL+128, *bc1L=vecL+256, *WaL=vecL+384, *Wc2L=vecL+512, *baL=vecL+640;

  const int tid = threadIdx.x;
  // ---- stage weights into fragment-swizzled LDS (once per WG) ----
  for (int i = tid; i < 272*HID; i += blockDim.x) {
    int k = i >> 7, n = i & 127;
    W1L[b_idx(k,n)] = (__bf16)W1g[i];
  }
  for (int i = tid; i < 16*HID; i += blockDim.x) {       // zero-pad K 272..287
    int k = 272 + (i >> 7), n = i & 127;
    W1L[b_idx(k,n)] = (__bf16)0.f;
  }
  for (int i = tid; i < HID*HID; i += blockDim.x) {
    int k = i >> 7, n = i & 127;
    W2L [b_idx(k,n)] = (__bf16)W2g[i];
    Wc1L[b_idx(k,n)] = (__bf16)Wc1g[i];
  }
  for (int i = tid; i < 128; i += blockDim.x) {
    b1L[i]=b1g[i]; b2L[i]=b2g[i]; bc1L[i]=bc1g[i]; WaL[i]=Wag[i]; Wc2L[i]=Wc2g[i];
  }
  if (tid == 0) baL[0] = bag[0];
  __syncthreads();

  const int wave = tid >> 5, lane = tid & 31;
  const int l = lane & 15, hi = lane >> 4;
  char* wbase = smem + OFF_WAVE + wave*PER_WAVE;
  __bf16* bufA = (__bf16*)(wbase + OFF_BUFA);
  __bf16* mbuf = (__bf16*)(wbase + OFF_MBUF);
  int*   srcb = (int*)(wbase + OFF_META);
  int*   dstb = srcb + 16;
  int*   valb = dstb + 16;
  float* cutb = (float*)(valb + 16);
  float* attb = cutb + 16;
  float* cwb  = attb + 16;
  float* evnb = cwb  + 16;   // 16x3

  const int tiles = (n_edges + 15) >> 4;
  for (int tile = blockIdx.x*WAVES + wave; tile < tiles; tile += gridDim.x*WAVES) {
    // ---- edge meta + RBF features (one edge per lane 0..15) ----
    if (lane < 16) {
      int e = tile*16 + lane;
      int valid = (e < n_edges) ? 1 : 0;
      int ec = valid ? e : 0;
      int sN = ei[ec], dN = ei[n_edges + ec];
      srcb[lane] = sN; dstb[lane] = dN; valb[lane] = valid;
      float dx = pos[3*dN]-pos[3*sN], dy = pos[3*dN+1]-pos[3*sN+1], dz = pos[3*dN+2]-pos[3*sN+2];
      float dist = sqrtf(dx*dx + dy*dy + dz*dz);
      float cut = (dist < CUTF) ? 0.5f*(__cosf(3.14159265358979f*dist/CUTF) + 1.f) : 0.f;
      cutb[lane] = cut;
      float inv = 1.f/(dist + 1e-8f);
      evnb[3*lane]=dx*inv; evnb[3*lane+1]=dy*inv; evnb[3*lane+2]=dz*inv;
      const float w = CUTF / (float)NRBF;
      #pragma unroll
      for (int j=0;j<NRBF;j++){
        float c = CUTF * (float)j / (float)(NRBF-1);
        float t = (dist - c)/w;
        bufA[a_idx(lane, 256+j)] = (__bf16)__expf(-t*t);
      }
      #pragma unroll
      for (int j=NRBF;j<32;j++) bufA[a_idx(lane, 256+j)] = (__bf16)0.f;
    }
    __builtin_amdgcn_wave_barrier();
    // ---- stage gathered h[src], h[dst] into A-fragment layout ----
    for (int idx = lane; idx < 16*HID; idx += 32) {
      int e = idx >> 7, k = idx & 127;
      bufA[a_idx(e, k)]       = (__bf16)h[(long)srcb[e]*HID + k];
      bufA[a_idx(e, 128 + k)] = (__bf16)h[(long)dstb[e]*HID + k];
    }
    __builtin_amdgcn_wave_barrier();

    // ---- GEMM1: (16x288) @ (288x128) ----
    v8f acc[8];
    #pragma unroll
    for (int t=0;t<8;t++){ v8f z = {0.f,0.f,0.f,0.f,0.f,0.f,0.f,0.f}; acc[t]=z; }
    #pragma unroll
    for (int s=0;s<9;s++){
      v16bf af = ((const v16bf*)(bufA + s*512))[lane];
      #pragma unroll
      for (int t=0;t<8;t++){
        v16bf bf = ((const v16bf*)(W1L + (s*8+t)*512))[lane];
        acc[t] = __builtin_amdgcn_wmma_f32_16x16x32_bf16(false, af, false, bf,
                                                         (short)0, acc[t], false, false);
      }
    }
    // bias + silu; store m1 (A-layout) for GEMM2
    #pragma unroll
    for (int t=0;t<8;t++){
      float bb = b1L[t*16 + l];
      #pragma unroll
      for (int r=0;r<8;r++){
        float x = fsilu(acc[t][r] + bb);
        mbuf[a_idx(r + hi*8, t*16 + l)] = (__bf16)x;
      }
    }
    __builtin_amdgcn_wave_barrier();

    // ---- GEMM2: (16x128) @ W2 ----
    #pragma unroll
    for (int t=0;t<8;t++){ v8f z = {0.f,0.f,0.f,0.f,0.f,0.f,0.f,0.f}; acc[t]=z; }
    #pragma unroll
    for (int s=0;s<4;s++){
      v16bf af = ((const v16bf*)(mbuf + s*512))[lane];
      #pragma unroll
      for (int t=0;t<8;t++){
        v16bf bf = ((const v16bf*)(W2L + (s*8+t)*512))[lane];
        acc[t] = __builtin_amdgcn_wmma_f32_16x16x32_bf16(false, af, false, bf,
                                                         (short)0, acc[t], false, false);
      }
    }
    #pragma unroll
    for (int t=0;t<8;t++){
      float bb = b2L[t*16 + l];
      #pragma unroll
      for (int r=0;r<8;r++) acc[t][r] = fsilu(acc[t][r] + bb);
    }

    // ---- attention: sigmoid(m @ Wa + ba) * cut (butterfly reduce over N) ----
    float p[8];
    #pragma unroll
    for (int r=0;r<8;r++) p[r] = 0.f;
    #pragma unroll
    for (int t=0;t<8;t++){
      float wa = WaL[t*16 + l];
      #pragma unroll
      for (int r=0;r<8;r++) p[r] += acc[t][r]*wa;
    }
    #pragma unroll
    for (int off=1; off<16; off<<=1){
      #pragma unroll
      for (int r=0;r<8;r++) p[r] += __shfl_xor(p[r], off, 32);
    }
    if (l == 0) {
      float ba0 = baL[0];
      #pragma unroll
      for (int r=0;r<8;r++){
        int m = r + hi*8;
        attb[m] = fsig(p[r] + ba0) * cutb[m] * (float)valb[m];
      }
    }
    __builtin_amdgcn_wave_barrier();

    // ---- scatter m*att into h-sum; also store m2 (A-layout) for GEMM3 ----
    #pragma unroll
    for (int t=0;t<8;t++){
      #pragma unroll
      for (int r=0;r<8;r++){
        int m = r + hi*8;
        float v = acc[t][r];
        mbuf[a_idx(m, t*16 + l)] = (__bf16)v;
        if (valb[m]) atomicAdd(&hsum[(long)dstb[m]*HID + t*16 + l], v*attb[m]);
      }
    }
    __builtin_amdgcn_wave_barrier();

    // ---- GEMM3: coord head silu(m@Wc1+bc1) @ Wc2 ----
    #pragma unroll
    for (int t=0;t<8;t++){ v8f z = {0.f,0.f,0.f,0.f,0.f,0.f,0.f,0.f}; acc[t]=z; }
    #pragma unroll
    for (int s=0;s<4;s++){
      v16bf af = ((const v16bf*)(mbuf + s*512))[lane];
      #pragma unroll
      for (int t=0;t<8;t++){
        v16bf bf = ((const v16bf*)(Wc1L + (s*8+t)*512))[lane];
        acc[t] = __builtin_amdgcn_wmma_f32_16x16x32_bf16(false, af, false, bf,
                                                         (short)0, acc[t], false, false);
      }
    }
    #pragma unroll
    for (int r=0;r<8;r++) p[r] = 0.f;
    #pragma unroll
    for (int t=0;t<8;t++){
      float bb = bc1L[t*16 + l], wc = Wc2L[t*16 + l];
      #pragma unroll
      for (int r=0;r<8;r++) p[r] += fsilu(acc[t][r] + bb) * wc;
    }
    #pragma unroll
    for (int off=1; off<16; off<<=1){
      #pragma unroll
      for (int r=0;r<8;r++) p[r] += __shfl_xor(p[r], off, 32);
    }
    if (l == 0) {
      #pragma unroll
      for (int r=0;r<8;r++){
        int m = r + hi*8;
        cwb[m] = p[r] * cutb[m] * (float)valb[m];
      }
    }
    __builtin_amdgcn_wave_barrier();
    // ---- scatter coordinate update ----
    if (lane < 16 && valb[lane]) {
      int dN = dstb[lane]; float c = cwb[lane];
      atomicAdd(&psum[(long)dN*3 + 0], evnb[3*lane + 0]*c);
      atomicAdd(&psum[(long)dN*3 + 1], evnb[3*lane + 1]*c);
      atomicAdd(&psum[(long)dN*3 + 2], evnb[3*lane + 2]*c);
    }
    __builtin_amdgcn_wave_barrier();
  }
}

__global__ void egnn_finalize_kernel(float* out, const float* __restrict__ pos,
                                     const float* __restrict__ deg, int n_nodes){
  long i = (long)blockIdx.x*blockDim.x + threadIdx.x;
  long stride = (long)gridDim.x*blockDim.x;
  long nh = (long)n_nodes*HID;
  for (long j=i; j<nh; j+=stride){
    float dn = deg[j >> 7]; dn = dn < 1.f ? 1.f : dn;
    out[j] = out[j] / dn;
  }
  long np = (long)n_nodes*3;
  float* po = out + nh;
  for (long j=i; j<np; j+=stride){
    float dn = deg[j/3]; dn = dn < 1.f ? 1.f : dn;
    po[j] = pos[j] + po[j]/dn;
  }
}

extern "C" void kernel_launch(void* const* d_in, const int* in_sizes, int n_in,
                              void* d_out, int out_size, void* d_ws, size_t ws_size,
                              hipStream_t stream) {
  const float* h   = (const float*)d_in[0];
  const float* pos = (const float*)d_in[1];
  const int*   ei  = (const int*)  d_in[2];
  const float* W1  = (const float*)d_in[3];
  const float* b1  = (const float*)d_in[4];
  const float* W2  = (const float*)d_in[5];
  const float* b2  = (const float*)d_in[6];
  const float* Wa  = (const float*)d_in[7];
  const float* ba  = (const float*)d_in[8];
  const float* Wc1 = (const float*)d_in[9];
  const float* bc1 = (const float*)d_in[10];
  const float* Wc2 = (const float*)d_in[11];

  int n_nodes = in_sizes[0] / HID;
  int n_edges = in_sizes[2] / 2;
  float* out  = (float*)d_out;
  float* deg  = (float*)d_ws;            // n_nodes floats
  float* hsum = out;                      // accumulate directly into output
  float* psum = out + (long)n_nodes*HID;

  egnn_zero_kernel<<<2048, 256, 0, stream>>>(out, (long)out_size, deg, n_nodes);
  egnn_degree_kernel<<<(n_edges + 255)/256, 256, 0, stream>>>(ei, deg, n_edges);

  int tiles = (n_edges + 15) / 16;
  int grid  = (tiles + WAVES - 1) / WAVES;
  if (grid > 1024) grid = 1024;
  egnn_edge_kernel<<<grid, 32*WAVES, SMEM_TOTAL, stream>>>(
      h, pos, ei, W1, b1, W2, b2, Wa, ba, Wc1, bc1, Wc2, hsum, psum, n_nodes, n_edges);

  egnn_finalize_kernel<<<2048, 256, 0, stream>>>(out, pos, deg, n_nodes);
}
